// DMPNN_33337536152098
// MI455X (gfx1250) — matrix operand order repaired and checked by reading, not compiled, not run on hardware
//
#include <hip/hip_runtime.h>

typedef __attribute__((ext_vector_type(16))) __bf16        v16bf;
typedef __attribute__((ext_vector_type(8)))  float         v8f;
typedef __attribute__((ext_vector_type(4)))  unsigned int  u32x4;
typedef __attribute__((ext_vector_type(4)))  int           i32x4;

#define AS1 __attribute__((address_space(1)))
#define AS3 __attribute__((address_space(3)))

#define DEVINL __device__ __forceinline__

#if defined(__has_builtin)
#if __has_builtin(__builtin_amdgcn_global_load_async_to_lds_b128)
#define HAVE_ASYNC_LDS 1
#endif
#if __has_builtin(__builtin_amdgcn_s_wait_asynccnt)
#define HAVE_WAIT_ASYNC 1
#endif
#endif

DEVINL float bf2f(unsigned short b) {
    union { unsigned u; float f; } c; c.u = ((unsigned)b) << 16; return c.f;
}
DEVINL unsigned short f2bf(float f) {
    union { float f; unsigned u; } c; c.f = f;
    unsigned u = c.u;
    return (unsigned short)((u + 0x7fffu + ((u >> 16) & 1u)) >> 16);
}

// ---------------------------------------------------------------- utilities
__global__ __launch_bounds__(256) void k_zero_u32(unsigned* __restrict__ p, long long n) {
    long long i = (long long)blockIdx.x * blockDim.x + threadIdx.x;
    if (i < n) p[i] = 0u;
}

__global__ __launch_bounds__(256) void k_cvt_bf16(const float* __restrict__ s,
                                                  unsigned short* __restrict__ d, long long n) {
    long long i = (long long)blockIdx.x * blockDim.x + threadIdx.x;
    if (i < n) d[i] = f2bf(s[i]);
}

// pack per-layer combined weight [W_w (256x160) | U_w (256x256)] -> 256x416 bf16, L=4 layers
__global__ __launch_bounds__(256) void k_pack_wc(const float* __restrict__ Ww,
                                                 const float* __restrict__ Uw,
                                                 unsigned short* __restrict__ d) {
    long long i = (long long)blockIdx.x * blockDim.x + threadIdx.x;
    const long long total = 4LL * 256 * 416;
    if (i >= total) return;
    int c = (int)(i % 416);
    int n = (int)((i / 416) % 256);
    int l = (int)(i / (416 * 256));
    float v = (c < 160) ? Ww[((long long)l * 256 + n) * 160 + c]
                        : Uw[((long long)l * 256 + n) * 256 + (c - 160)];
    d[i] = f2bf(v);
}

// feats[e, 0:128] = x[tgt[e]] ; feats[e, 128:160] = edge_attr[e]
__global__ __launch_bounds__(256) void k_feats(const float* __restrict__ x,
                                               const float* __restrict__ ea,
                                               const int* __restrict__ tgt,
                                               unsigned short* __restrict__ feats, int E) {
    long long i = (long long)blockIdx.x * blockDim.x + threadIdx.x;
    if (i >= (long long)E * 160) return;
    int e = (int)(i / 160);
    int c = (int)(i % 160);
    float v = (c < 128) ? x[(long long)tgt[e] * 128 + c] : ea[(long long)e * 32 + (c - 128)];
    feats[i] = f2bf(v);
}

// s_uv[e] = sum_j msg_pad[nbr[e][j]] ; 2 channels per thread
__global__ __launch_bounds__(256) void k_suv(const unsigned short* __restrict__ msg_pad,
                                             const int* __restrict__ nbr,
                                             unsigned short* __restrict__ s_uv, int E) {
    long long i = (long long)blockIdx.x * blockDim.x + threadIdx.x;
    if (i >= (long long)E * 128) return;
    int e = (int)(i >> 7);
    int hp = (int)(i & 127) << 1;
    float a0 = 0.f, a1 = 0.f;
#pragma unroll
    for (int j = 0; j < 3; ++j) {
        int n = nbr[e * 3 + j];
        unsigned w = *(const unsigned*)(msg_pad + (long long)n * 256 + hp);
        a0 += bf2f((unsigned short)(w & 0xffffu));
        a1 += bf2f((unsigned short)(w >> 16));
    }
    unsigned r = (unsigned)f2bf(a0) | ((unsigned)f2bf(a1) << 16);
    *(unsigned*)(s_uv + (long long)e * 256 + hp) = r;
}

// r_dash[e] = sum_j r_pad[nbr[e][j]] * msg_pad[nbr[e][j]]
__global__ __launch_bounds__(256) void k_rdash(const unsigned short* __restrict__ msg_pad,
                                               const unsigned short* __restrict__ r_pad,
                                               const int* __restrict__ nbr,
                                               unsigned short* __restrict__ r_dash, int E) {
    long long i = (long long)blockIdx.x * blockDim.x + threadIdx.x;
    if (i >= (long long)E * 128) return;
    int e = (int)(i >> 7);
    int hp = (int)(i & 127) << 1;
    float a0 = 0.f, a1 = 0.f;
#pragma unroll
    for (int j = 0; j < 3; ++j) {
        int n = nbr[e * 3 + j];
        unsigned wm = *(const unsigned*)(msg_pad + (long long)n * 256 + hp);
        unsigned wr = *(const unsigned*)(r_pad + (long long)n * 256 + hp);
        a0 += bf2f((unsigned short)(wm & 0xffffu)) * bf2f((unsigned short)(wr & 0xffffu));
        a1 += bf2f((unsigned short)(wm >> 16)) * bf2f((unsigned short)(wr >> 16));
    }
    unsigned r = (unsigned)f2bf(a0) | ((unsigned)f2bf(a1) << 16);
    *(unsigned*)(r_dash + (long long)e * 256 + hp) = r;
}

// aggr[tgt[e]] += msg[e]
__global__ __launch_bounds__(256) void k_scatter(const unsigned short* __restrict__ msg_pad,
                                                 const int* __restrict__ tgt,
                                                 float* __restrict__ aggr, int E) {
    long long i = (long long)blockIdx.x * blockDim.x + threadIdx.x;
    if (i >= (long long)E * 256) return;
    int e = (int)(i >> 8);
    int h = (int)(i & 255);
    atomicAdd(&aggr[(long long)tgt[e] * 256 + h], bf2f(msg_pad[(long long)(e + 1) * 256 + h]));
}

// x_edge[he] = [ |xg_u - xg_v| , xg_u + xg_v ]
__global__ __launch_bounds__(256) void k_edgefeat(const unsigned short* __restrict__ xg,
                                                  const int* __restrict__ ei, int E, int Eh,
                                                  unsigned short* __restrict__ xe) {
    long long i = (long long)blockIdx.x * blockDim.x + threadIdx.x;
    if (i >= (long long)Eh * 256) return;
    int he = (int)(i >> 8);
    int c = (int)(i & 255);
    int u = ei[2 * he];
    int v = ei[E + 2 * he];
    float a = bf2f(xg[(long long)u * 256 + c]);
    float b = bf2f(xg[(long long)v * 256 + c]);
    xe[(long long)he * 512 + c]       = f2bf(fabsf(a - b));
    xe[(long long)he * 512 + 256 + c] = f2bf(a + b);
}

// logits[row] = dot(Hb[row, 0:512], w) + b[0]   (one wave32 per row)
__global__ __launch_bounds__(256) void k_dot512(const unsigned short* __restrict__ Hb,
                                                const float* __restrict__ w,
                                                const float* __restrict__ bptr,
                                                float* __restrict__ out, int rows) {
    int wid  = (int)((blockIdx.x * (long long)blockDim.x + threadIdx.x) >> 5);
    int lane = threadIdx.x & 31;
    if (wid >= rows) return;
    const unsigned short* hp = Hb + (long long)wid * 512;
    float s = 0.f;
#pragma unroll
    for (int k = lane; k < 512; k += 32) s += bf2f(hp[k]) * w[k];
#pragma unroll
    for (int off = 16; off > 0; off >>= 1) s += __shfl_xor(s, off, 32);
    if (lane == 0) out[wid] = s + bptr[0];
}

// -------- two-stage online softmax reduction
__global__ __launch_bounds__(256) void k_softmax_part(const float* __restrict__ lg, int n,
                                                      float* __restrict__ pm,
                                                      float* __restrict__ ps) {
    __shared__ float sh[256];
    const int stride = 256 * gridDim.x;
    float m = -3.4e38f;
    for (int i = blockIdx.x * 256 + threadIdx.x; i < n; i += stride) m = fmaxf(m, lg[i]);
    sh[threadIdx.x] = m; __syncthreads();
    for (int s = 128; s > 0; s >>= 1) {
        if (threadIdx.x < (unsigned)s) sh[threadIdx.x] = fmaxf(sh[threadIdx.x], sh[threadIdx.x + s]);
        __syncthreads();
    }
    float mx = sh[0]; __syncthreads();
    float su = 0.f;
    for (int i = blockIdx.x * 256 + threadIdx.x; i < n; i += stride) su += expf(lg[i] - mx);
    sh[threadIdx.x] = su; __syncthreads();
    for (int s = 128; s > 0; s >>= 1) {
        if (threadIdx.x < (unsigned)s) sh[threadIdx.x] += sh[threadIdx.x + s];
        __syncthreads();
    }
    if (threadIdx.x == 0) { pm[blockIdx.x] = mx; ps[blockIdx.x] = sh[0]; }
}

__global__ __launch_bounds__(256) void k_softmax_comb(const float* __restrict__ pm,
                                                      const float* __restrict__ ps, int nb,
                                                      float* __restrict__ stats) {
    __shared__ float sh[256];
    float m = -3.4e38f;
    for (int i = threadIdx.x; i < nb; i += 256) m = fmaxf(m, pm[i]);
    sh[threadIdx.x] = m; __syncthreads();
    for (int s = 128; s > 0; s >>= 1) {
        if (threadIdx.x < (unsigned)s) sh[threadIdx.x] = fmaxf(sh[threadIdx.x], sh[threadIdx.x + s]);
        __syncthreads();
    }
    float M = sh[0]; __syncthreads();
    float su = 0.f;
    for (int i = threadIdx.x; i < nb; i += 256) su += ps[i] * expf(pm[i] - M);
    sh[threadIdx.x] = su; __syncthreads();
    for (int s = 128; s > 0; s >>= 1) {
        if (threadIdx.x < (unsigned)s) sh[threadIdx.x] += sh[threadIdx.x + s];
        __syncthreads();
    }
    if (threadIdx.x == 0) { stats[0] = M; stats[1] = sh[0]; }
}

__global__ __launch_bounds__(256) void k_softmax_norm(const float* __restrict__ lg,
                                                      const float* __restrict__ stats,
                                                      float* __restrict__ out, int n) {
    long long i = (long long)blockIdx.x * blockDim.x + threadIdx.x;
    if (i < n) out[i] = expf(lg[i] - stats[0]) / stats[1];
}

// ---------------------------------------------------------------- WMMA GEMM
// OUT[M x Nout] = epi( concat(A1[:, :SPLIT*32], A2) @ W^T + bias )
// A1/A2/W/out are bf16 (raw u16 bits), W row-major Nout x (NK*32).
// Block: 256 threads = 8 waves; wave -> 32x64 output strip (2 row-tiles x 4 col-tiles):
// each LDS B fragment feeds 2 WMMAs, halving LDS traffic per FLOP.
// W slice (64 rows x KT) staged in LDS (row pad +8 elem -> conflict-free b128 reads).
// NK/SPLIT compile-time: fully unrolled k-loop, base+immediate load addresses.
enum { EPI_SIG = 0, EPI_RELU = 1, EPI_COMBINE = 2 };

struct Frag {
    union { v16bf v; u32x4 q[2]; };
};

DEVINL void load_afrag(Frag& a, const unsigned short* b1h, const unsigned short* b2h,
                       int ks, int split) {
    const unsigned short* p = (ks < split) ? (b1h + ks * 32) : (b2h + (ks - split) * 32);
    const u32x4* q = (const u32x4*)p;
    a.q[0] = q[0];   // K +0..7  (or +8..15)
    a.q[1] = q[2];   // K +16..23 (or +24..31)
}

template <int EPI, int NK, int SPLIT>
__global__ __launch_bounds__(256) void gemm_ws(
        const unsigned short* __restrict__ A1, int lda1,
        const unsigned short* __restrict__ A2, int lda2,
        const unsigned short* __restrict__ W,
        const float* __restrict__ bias,
        unsigned short* __restrict__ out, int M, int Nout,
        const unsigned short* __restrict__ zbuf,
        const unsigned short* __restrict__ sbuf) {
    constexpr int KT   = NK * 32;
    constexpr int LDSS = KT + 8;
    extern __shared__ unsigned short smem[];   // 64 x LDSS
    const int lane    = threadIdx.x & 31;
    const int wave    = threadIdx.x >> 5;
    const int rowBase = blockIdx.x * 256 + wave * 32;
    const int n0base  = blockIdx.y * 64;
    const int halfSel = lane >> 4;   // 0: lanes 0-15, 1: lanes 16-31
    const int l15     = lane & 15;

    // ---- stage W slice [n0base, n0base+64) x KT into LDS
    {
        constexpr int cpr = KT >> 3;               // 16B chunks per row
#pragma unroll 4
        for (int c = threadIdx.x; c < 64 * cpr; c += 256) {
            int row = c / cpr;
            int kk  = (c - row * cpr) << 3;
            const unsigned short* src = W + (long long)(n0base + row) * KT + kk;
            unsigned short*       dst = smem + row * LDSS + kk;
#if defined(HAVE_ASYNC_LDS)
            __builtin_amdgcn_global_load_async_to_lds_b128(
                (AS1 i32x4*)(void*)src, (AS3 i32x4*)(void*)dst, 0, 0);
#else
            *(u32x4*)dst = *(const u32x4*)src;
#endif
        }
#if defined(HAVE_ASYNC_LDS)
#if defined(HAVE_WAIT_ASYNC)
        __builtin_amdgcn_s_wait_asynccnt(0);
#else
        asm volatile("s_wait_asynccnt 0" ::: "memory");
#endif
#endif
    }
    __syncthreads();

    // ---- two clamped A rows (OOB rows compute garbage, never stored)
    int m0 = rowBase + l15;        if (m0 >= M) m0 = M - 1;
    int m1 = rowBase + 16 + l15;   if (m1 >= M) m1 = M - 1;
    const unsigned short* b1h0 = A1 + (long long)m0 * lda1 + halfSel * 8;
    const unsigned short* b2h0 = A2 + (long long)m0 * lda2 + halfSel * 8;
    const unsigned short* b1h1 = A1 + (long long)m1 * lda1 + halfSel * 8;
    const unsigned short* b2h1 = A2 + (long long)m1 * lda2 + halfSel * 8;

    // per-tile LDS B base pointers (lane l15 -> column n0base + t*16 + l15)
    const unsigned short* s0 = smem + (0 * 16 + l15) * LDSS + halfSel * 8;
    const unsigned short* s1 = smem + (1 * 16 + l15) * LDSS + halfSel * 8;
    const unsigned short* s2 = smem + (2 * 16 + l15) * LDSS + halfSel * 8;
    const unsigned short* s3 = smem + (3 * 16 + l15) * LDSS + halfSel * 8;

    // acc[t*2 + rr]: col-tile t (0..3), row-tile rr (0..1)
    v8f acc[8];
#pragma unroll
    for (int t = 0; t < 4; ++t) {
        float bc = bias[n0base + t * 16 + l15];
#pragma unroll
        for (int i = 0; i < 8; ++i) { acc[t * 2][i] = bc; acc[t * 2 + 1][i] = bc; }
    }

    // ---- fully unrolled k loop: all offsets are immediates, no phis
#pragma unroll
    for (int ks = 0; ks < NK; ++ks) {
        Frag a0, a1;
        load_afrag(a0, b1h0, b2h0, ks, SPLIT);
        load_afrag(a1, b1h1, b2h1, ks, SPLIT);
        Frag b[4];
        {
            const int ko = ks * 32;
            b[0].q[0] = *(const u32x4*)(s0 + ko); b[0].q[1] = *(const u32x4*)(s0 + ko + 16);
            b[1].q[0] = *(const u32x4*)(s1 + ko); b[1].q[1] = *(const u32x4*)(s1 + ko + 16);
            b[2].q[0] = *(const u32x4*)(s2 + ko); b[2].q[1] = *(const u32x4*)(s2 + ko + 16);
            b[3].q[0] = *(const u32x4*)(s3 + ko); b[3].q[1] = *(const u32x4*)(s3 + ko + 16);
        }
#pragma unroll
        for (int t = 0; t < 4; ++t) {
            acc[t * 2]     = __builtin_amdgcn_wmma_f32_16x16x32_bf16(
                false, a0.v, false, b[t].v, (short)0, acc[t * 2], false, false);
            acc[t * 2 + 1] = __builtin_amdgcn_wmma_f32_16x16x32_bf16(
                false, a1.v, false, b[t].v, (short)0, acc[t * 2 + 1], false, false);
        }
    }

    // ---- epilogue: lane -> col = n0 + l15, VGPR r -> row = rowBase + rr*16 + halfSel*8 + r
#pragma unroll
    for (int t = 0; t < 4; ++t) {
        int col = n0base + t * 16 + l15;
#pragma unroll
        for (int rr = 0; rr < 2; ++rr) {
#pragma unroll
            for (int r = 0; r < 8; ++r) {
                int row = rowBase + rr * 16 + halfSel * 8 + r;
                if (row >= M) continue;
                float v = acc[t * 2 + rr][r];
                if (EPI == EPI_SIG) {
                    v = 1.f / (1.f + expf(-v));
                } else if (EPI == EPI_RELU) {
                    v = v > 0.f ? v : 0.f;
                } else { // EPI_COMBINE: new_msg = (1-z)*s_uv + z*tanh(v)
                    float md = tanhf(v);
                    float zz = bf2f(zbuf[(long long)row * 256 + col]);
                    float ss = bf2f(sbuf[(long long)row * 256 + col]);
                    v = (1.f - zz) * ss + zz * md;
                }
                out[(long long)row * Nout + col] = f2bf(v);
            }
        }
    }
}

// ---------------------------------------------------------------- host
static inline unsigned gsz(long long n, int b) { return (unsigned)((n + b - 1) / b); }
static inline size_t smem_bytes(int ktot) { return (size_t)64 * (ktot + 8) * 2; }

extern "C" void kernel_launch(void* const* d_in, const int* in_sizes, int n_in,
                              void* d_out, int out_size, void* d_ws, size_t ws_size,
                              hipStream_t stream) {
    const float* x    = (const float*)d_in[0];
    const float* ea   = (const float*)d_in[1];
    const int*   ei   = (const int*)d_in[2];     // (2,E) row-major
    const int*   nbr  = (const int*)d_in[3];     // (E,3)
    const float* Wz_w = (const float*)d_in[4];
    const float* Wz_b = (const float*)d_in[5];
    const float* Wr_w = (const float*)d_in[6];
    const float* Wr_b = (const float*)d_in[7];
    const float* W_w  = (const float*)d_in[8];
    const float* U_w  = (const float*)d_in[9];
    const float* U_b  = (const float*)d_in[10];
    const float* mlp_w = (const float*)d_in[11];
    const float* mlp_b = (const float*)d_in[12];
    const float* nc1_w = (const float*)d_in[13];
    const float* nc1_b = (const float*)d_in[14];
    const float* nc2_w = (const float*)d_in[15];
    const float* nc2_b = (const float*)d_in[16];
    const float* ec1_w = (const float*)d_in[17];
    const float* ec1_b = (const float*)d_in[18];
    const float* ec2_w = (const float*)d_in[19];
    const float* ec2_b = (const float*)d_in[20];

    const int N  = in_sizes[0] / 128;   // 60000
    const int E  = in_sizes[1] / 32;    // 119998
    const int Eh = E / 2;               // 59999
    const int* tgt = ei + E;

    // -------- workspace carve-up (256B aligned)
    size_t off = 0;
    auto carve = [&](size_t bytes) -> char* {
        off = (off + 255) & ~(size_t)255;
        char* p = (char*)d_ws + off;
        off += bytes;
        return p;
    };
    unsigned short* wzB   = (unsigned short*)carve(4LL * 256 * 416 * 2);
    unsigned short* wrB   = (unsigned short*)carve(4LL * 256 * 416 * 2);
    unsigned short* wcB   = (unsigned short*)carve(4LL * 256 * 416 * 2);
    unsigned short* mlpB  = (unsigned short*)carve(256LL * 384 * 2);
    unsigned short* nc1B  = (unsigned short*)carve(512LL * 256 * 2);
    unsigned short* ec1B  = (unsigned short*)carve(512LL * 512 * 2);
    unsigned short* feats = (unsigned short*)carve((long long)E * 160 * 2);
    unsigned short* xB    = (unsigned short*)carve((long long)N * 128 * 2);
    unsigned short* msgP  = (unsigned short*)carve((long long)(E + 2) * 256 * 2);
    unsigned short* rP    = (unsigned short*)carve((long long)(E + 2) * 256 * 2);
    unsigned short* sUV   = (unsigned short*)carve((long long)(E + 2) * 256 * 2);
    unsigned short* rD    = (unsigned short*)carve((long long)(E + 2) * 256 * 2);
    unsigned short* zB    = (unsigned short*)carve((long long)(E + 2) * 256 * 2);
    float*          aggrF = (float*)carve((long long)N * 256 * 4);
    unsigned short* aggrB = (unsigned short*)carve((long long)N * 256 * 2);
    unsigned short* xg    = (unsigned short*)carve((long long)N * 256 * 2);
    unsigned short* h1    = (unsigned short*)carve((long long)N * 512 * 2);
    unsigned short* xE    = (unsigned short*)carve((long long)Eh * 512 * 2);
    float*          lgts  = (float*)carve((long long)(N + Eh) * 4);
    float*          pm    = (float*)carve(512 * 4);
    float*          ps    = (float*)carve(512 * 4);
    float*          stats = (float*)carve(64);
    (void)ws_size; (void)n_in;

    // -------- per-call init (graph-replay safe: everything re-derived)
    k_zero_u32<<<gsz((long long)(E + 2) * 128, 256), 256, 0, stream>>>((unsigned*)msgP, (long long)(E + 2) * 128);
    k_zero_u32<<<1, 256, 0, stream>>>((unsigned*)rP, 128);                 // r_pad row 0
    k_zero_u32<<<gsz((long long)N * 256, 256), 256, 0, stream>>>((unsigned*)aggrF, (long long)N * 256);

    // weights -> bf16
    k_cvt_bf16<<<gsz(4LL * 256 * 416, 256), 256, 0, stream>>>(Wz_w, wzB, 4LL * 256 * 416);
    k_cvt_bf16<<<gsz(4LL * 256 * 416, 256), 256, 0, stream>>>(Wr_w, wrB, 4LL * 256 * 416);
    k_pack_wc<<<gsz(4LL * 256 * 416, 256), 256, 0, stream>>>(W_w, U_w, wcB);
    k_cvt_bf16<<<gsz(256LL * 384, 256), 256, 0, stream>>>(mlp_w, mlpB, 256LL * 384);
    k_cvt_bf16<<<gsz(512LL * 256, 256), 256, 0, stream>>>(nc1_w, nc1B, 512LL * 256);
    k_cvt_bf16<<<gsz(512LL * 512, 256), 256, 0, stream>>>(ec1_w, ec1B, 512LL * 512);
    k_cvt_bf16<<<gsz((long long)N * 128, 256), 256, 0, stream>>>(x, xB, (long long)N * 128);
    k_feats<<<gsz((long long)E * 160, 256), 256, 0, stream>>>(x, ea, tgt, feats, E);

    // -------- message-passing layers (NK=13, SPLIT=5)
    dim3 gE(gsz(E, 256), 4);   // Nout = 256, 256 rows per block
    const size_t sm416 = smem_bytes(416);
    for (int l = 0; l < 4; ++l) {
        const unsigned short* wz = wzB + (long long)l * 256 * 416;
        const unsigned short* wr = wrB + (long long)l * 256 * 416;
        const unsigned short* wc = wcB + (long long)l * 256 * 416;
        k_suv<<<gsz((long long)E * 128, 256), 256, 0, stream>>>(msgP, nbr, sUV, E);
        gemm_ws<EPI_SIG, 13, 5><<<gE, 256, sm416, stream>>>(feats, 160, sUV, 256, wz,
                                                            Wz_b + l * 256, zB, E, 256,
                                                            nullptr, nullptr);
        gemm_ws<EPI_SIG, 13, 5><<<gE, 256, sm416, stream>>>(feats, 160, msgP + 256, 256, wr,
                                                            Wr_b + l * 256, rP + 256, E, 256,
                                                            nullptr, nullptr);
        k_rdash<<<gsz((long long)E * 128, 256), 256, 0, stream>>>(msgP, rP, nbr, rD, E);
        gemm_ws<EPI_COMBINE, 13, 5><<<gE, 256, sm416, stream>>>(feats, 160, rD, 256, wc,
                                                                U_b + l * 256, msgP + 256, E, 256,
                                                                zB, sUV);
    }

    // -------- readout
    k_scatter<<<gsz((long long)E * 256, 256), 256, 0, stream>>>(msgP, tgt, aggrF, E);
    k_cvt_bf16<<<gsz((long long)N * 256, 256), 256, 0, stream>>>(aggrF, aggrB, (long long)N * 256);

    dim3 gN4(gsz(N, 256), 4);
    gemm_ws<EPI_RELU, 12, 4><<<gN4, 256, smem_bytes(384), stream>>>(xB, 128, aggrB, 256, mlpB,
                                                                    mlp_b, xg, N, 256,
                                                                    nullptr, nullptr);
    dim3 gN8(gsz(N, 256), 8);
    gemm_ws<EPI_RELU, 8, 8><<<gN8, 256, smem_bytes(256), stream>>>(xg, 256, xg, 256, nc1B,
                                                                   nc1_b, h1, N, 512,
                                                                   nullptr, nullptr);
    k_dot512<<<gsz((long long)N * 32, 256), 256, 0, stream>>>(h1, nc2_w, nc2_b, lgts, N);

    k_edgefeat<<<gsz((long long)Eh * 256, 256), 256, 0, stream>>>(xg, ei, E, Eh, xE);
    dim3 gEh8(gsz(Eh, 256), 8);
    gemm_ws<EPI_RELU, 16, 16><<<gEh8, 256, smem_bytes(512), stream>>>(xE, 512, xE, 512, ec1B,
                                                                      ec1_b, h1, Eh, 512,
                                                                      nullptr, nullptr);
    k_dot512<<<gsz((long long)Eh * 32, 256), 256, 0, stream>>>(h1, ec2_w, ec2_b, lgts + N, Eh);

    // -------- global softmax over [node_logits ; edge_logits]
    const int ntot = N + Eh;
    k_softmax_part<<<512, 256, 0, stream>>>(lgts, ntot, pm, ps);
    k_softmax_comb<<<1, 256, 0, stream>>>(pm, ps, 512, stats);
    k_softmax_norm<<<gsz(ntot, 256), 256, 0, stream>>>(lgts, stats, (float*)d_out, ntot);
    (void)out_size;
}
